// LocalInteraction_75282186764963
// MI455X (gfx1250) — matrix operand order repaired and checked by reading, not compiled, not run on hardware
//
#include <hip/hip_runtime.h>
#include <hip/hip_bf16.h>

typedef unsigned short u16;
typedef unsigned int   u32;
typedef __attribute__((ext_vector_type(16))) __bf16 v16bf;
typedef __attribute__((ext_vector_type(8)))  __bf16 v8bf;
typedef __attribute__((ext_vector_type(2)))  __bf16 bf16x2;
typedef __attribute__((ext_vector_type(8)))  float  v8f;

#define WAVES_PER_WG 7
#define WAVE_LDS_U16 22656                      /* 45312 B per wave */
#define SMEM_BYTES (WAVES_PER_WG * WAVE_LDS_U16 * 2) /* 317184 B */

#define LDS_FENCE() asm volatile("" ::: "memory")

// ---------- conversions (native bf16 cvt hardware) ----------
__device__ __forceinline__ u16 f2bf(float f) {
  union { __bf16 h; u16 u; } c;
  c.h = (__bf16)f;
  return c.u;
}
__device__ __forceinline__ u32 f2bf2(float a, float b) {  // packed pair
  union { bf16x2 v; u32 u; } c;
  c.v = bf16x2{(__bf16)a, (__bf16)b};
  return c.u;
}
__device__ __forceinline__ float bf2f(u16 h) {
  return __uint_as_float(((u32)h) << 16);
}
__device__ __forceinline__ float bflo(u32 p) { return __uint_as_float(p << 16); }
__device__ __forceinline__ float bfhi(u32 p) { return __uint_as_float(p & 0xffff0000u); }

__device__ __forceinline__ v8f wmma_bf16(v16bf a, v16bf b, v8f c) {
  return __builtin_amdgcn_wmma_f32_16x16x32_bf16(false, a, false, b, (short)0, c,
                                                 false, false);
}

// A-fragment (16x32 bf16) from a row-major matrix in LDS: one 32B run.
__device__ __forceinline__ v16bf loadAfrag(const u16* rm, int rowStride, int mt,
                                           int kt, int lane) {
  int m = mt * 16 + (lane & 15);
  int off = kt * 32 + ((lane >> 4) << 4);
  const u16* p = rm + m * rowStride + off;
  union { v16bf v; v8bf h[2]; } u;
  u.h[0] = *(const v8bf*)(p);
  u.h[1] = *(const v8bf*)(p + 8);
  return u.v;
}
// B-fragment (32x16) from a matrix stored row-major as [n][k]: one 32B run.
__device__ __forceinline__ v16bf loadBfrag(const u16* nm, int rowStride, int kt,
                                           int ct, int lane) {
  int n = ct * 16 + (lane & 15);
  int off = kt * 32 + ((lane >> 4) << 4);
  const u16* p = nm + n * rowStride + off;
  union { v16bf v; v8bf h[2]; } u;
  u.h[0] = *(const v8bf*)(p);
  u.h[1] = *(const v8bf*)(p + 8);
  return u.v;
}
// pre-packed B-fragment straight from global (L2-resident weights)
__device__ __forceinline__ v16bf loadBfragG(const u16* g, int frag, int lane) {
  return *(const v16bf*)(g + frag * 512 + lane * 16);
}

// ---------- weight pre-pack kernels ----------
__global__ void pack_proj_kernel(const float* __restrict__ w_af,
                                 const float* __restrict__ w_df,
                                 const float* __restrict__ w_wf,
                                 const float* __restrict__ w_sf,
                                 u16* __restrict__ Bproj) {
  int idx = blockIdx.x * blockDim.x + threadIdx.x;  // 65536
  int e = idx & 511, ct = (idx >> 9) & 15, kt = idx >> 13;
  int lane = e >> 4, slot = e & 15;
  int nloc = lane & 15, khalf = lane >> 4;
  int c = kt * 32 + khalf * 16 + slot;
  int o = ct * 16 + nloc;
  int s = o >> 6, d = o & 63;
  const float* w = (s == 0) ? w_af : (s == 1) ? w_df : (s == 2) ? w_wf : w_sf;
  Bproj[idx] = f2bf(w[d * 256 + c]);
}

__global__ void pack_qkvo_kernel(const float* __restrict__ wq,
                                 const float* __restrict__ wk,
                                 const float* __restrict__ wv,
                                 const float* __restrict__ wo,
                                 u16* __restrict__ Bqkvo) {
  int idx = blockIdx.x * blockDim.x + threadIdx.x;  // 16384
  int mat = idx >> 12;
  int r = idx & 4095;
  int kt = r >> 11, ct = (r >> 9) & 3, e = r & 511;
  int lane = e >> 4, slot = e & 15;
  int nloc = lane & 15, khalf = lane >> 4;
  int k = kt * 32 + khalf * 16 + slot;
  int n = ct * 16 + nloc;
  const float* W = (mat == 0) ? wq : (mat == 1) ? wk : (mat == 2) ? wv : wo;
  Bqkvo[idx] = f2bf(W[n * 64 + k]);
}

__global__ void pack_small_kernel(const float* __restrict__ b_af, const float* __restrict__ b_df,
                                  const float* __restrict__ b_wf, const float* __restrict__ b_sf,
                                  const float* __restrict__ w_angle, const float* __restrict__ w_depth,
                                  const float* __restrict__ w_width, const float* __restrict__ w_score,
                                  const float* __restrict__ b_angle, const float* __restrict__ b_depth,
                                  const float* __restrict__ b_width, const float* __restrict__ b_score,
                                  float* __restrict__ bias_all, float* __restrict__ Whead,
                                  float* __restrict__ bhead) {
  int i = blockIdx.x * blockDim.x + threadIdx.x;
  if (i < 256) {
    int s = i >> 6, d = i & 63;
    const float* bs = (s == 0) ? b_af : (s == 1) ? b_df : (s == 2) ? b_wf : b_sf;
    bias_all[i] = bs[d];
  } else if (i < 256 + 1600) {
    int j = i - 256, orow = j >> 6, d = j & 63;
    float v;
    if (orow < 13)      v = w_angle[orow * 64 + d];
    else if (orow < 18) v = w_depth[(orow - 13) * 64 + d];
    else if (orow < 19) v = w_width[d];
    else                v = w_score[(orow - 19) * 64 + d];
    Whead[j] = v;
  } else if (i < 256 + 1600 + 25) {
    int orow = i - 1856;
    float v;
    if (orow < 13)      v = b_angle[orow];
    else if (orow < 18) v = b_depth[orow - 13];
    else if (orow < 19) v = b_width[0];
    else                v = b_score[orow - 19];
    bhead[orow] = v;
  }
}

// ---------- fused main kernel ----------
__global__ __launch_bounds__(224) void li_main_kernel(
    const float* __restrict__ vp,
    const u16* __restrict__ Bproj, const u16* __restrict__ Bqkvo,
    const float* __restrict__ bias_all,
    const float* __restrict__ bq, const float* __restrict__ bk,
    const float* __restrict__ bv, const float* __restrict__ bo,
    const float* __restrict__ ln_g, const float* __restrict__ ln_b,
    const float* __restrict__ Whead, const float* __restrict__ bhead,
    float* __restrict__ out) {
  extern __shared__ u16 smem[];
  const int lane = threadIdx.x & 31;
  const int wave = threadIdx.x >> 5;
  u16* base = smem + wave * WAVE_LDS_U16;
  u16* xRM = base;          // X tile 16 x 264 bf16; overlay: scores f32
  u16* tRM = base + 4224;   // tokens 64 x 72 (row-major)
  u16* qRM = base + 8832;   // Q rows;   overlay: msa rows (stage 6-7)
  u16* kRM = base + 13440;  // K rows;   overlay: probs, then x(bf16)
  u16* vT  = base + 18048;  // V transposed [d][token]
  float* scF = (float*)xRM; // 64 rows x 17 f32 scores

  const int g = blockIdx.x * WAVES_PER_WG + wave;  // tile id
  if (g >= 4096) return;
  const int b = g >> 8;
  const int n0 = (g & 255) << 4;
  const float* xb = vp + (size_t)b * 256 * 4096;

  // ---- stage 1: global f32 X tile -> bf16 rows (packed cvt + b32 writes) ----
  for (int it = 0; it < 16; ++it) {
    int idx = it * 32 + lane;
    int cp = idx >> 2, mg = idx & 3;
    int c0 = cp * 2;
    const float4 va = *(const float4*)(xb + (size_t)c0 * 4096 + n0 + mg * 4);
    const float4 vb = *(const float4*)(xb + (size_t)(c0 + 1) * 4096 + n0 + mg * 4);
    float a4[4] = {va.x, va.y, va.z, va.w};
    float b4[4] = {vb.x, vb.y, vb.z, vb.w};
#pragma unroll
    for (int j = 0; j < 4; ++j)
      *(u32*)(xRM + (mg * 4 + j) * 264 + c0) = f2bf2(a4[j], b4[j]);
  }
  LDS_FENCE();

  // ---- stage 2: projection; X A-fragments resident in 64 VGPRs ----
  {
    v16bf aX[8];
#pragma unroll
    for (int kt = 0; kt < 8; ++kt) aX[kt] = loadAfrag(xRM, 264, 0, kt, lane);
    for (int ct = 0; ct < 16; ++ct) {
      float bias = bias_all[ct * 16 + (lane & 15)];
      v8f acc;
#pragma unroll
      for (int r = 0; r < 8; ++r) acc[r] = bias;
#pragma unroll
      for (int kt = 0; kt < 8; ++kt)
        acc = wmma_bf16(aX[kt], loadBfragG(Bproj, kt * 16 + ct, lane), acc);
      int s = ct >> 2;
      int d = (ct & 3) * 16 + (lane & 15);
#pragma unroll
      for (int r = 0; r < 8; ++r) {
        int m = r + 8 * (lane >> 4);
        tRM[(m * 4 + s) * 72 + d] = f2bf(acc[r]);
      }
    }
  }
  LDS_FENCE();

  // ---- stage 3: q, k, v linears; token A-fragments resident ----
  {
    v16bf aT[8];
#pragma unroll
    for (int mt = 0; mt < 4; ++mt)
#pragma unroll
      for (int kt = 0; kt < 2; ++kt)
        aT[mt * 2 + kt] = loadAfrag(tRM, 72, mt, kt, lane);

    for (int which = 0; which < 3; ++which) {
      const u16* Bw = Bqkvo + which * 4096;
      const float* bp = (which == 0) ? bq : (which == 1) ? bk : bv;
      for (int ct = 0; ct < 4; ++ct) {
        v16bf bF0 = loadBfragG(Bw, ct, lane);       // L2-hot, loaded once
        v16bf bF1 = loadBfragG(Bw, 4 + ct, lane);
        float bias = bp[ct * 16 + (lane & 15)];
        int d = ct * 16 + (lane & 15);
#pragma unroll
        for (int mt = 0; mt < 4; ++mt) {
          v8f acc;
#pragma unroll
          for (int r = 0; r < 8; ++r) acc[r] = bias;
          acc = wmma_bf16(aT[mt * 2], bF0, acc);
          acc = wmma_bf16(aT[mt * 2 + 1], bF1, acc);
          int mb = mt * 16 + 8 * (lane >> 4);
          if (which == 0) {
#pragma unroll
            for (int r = 0; r < 8; ++r) qRM[(mb + r) * 72 + d] = f2bf(acc[r]);
          } else if (which == 1) {
#pragma unroll
            for (int r = 0; r < 8; ++r) kRM[(mb + r) * 72 + d] = f2bf(acc[r]);
          } else {
#pragma unroll
            for (int r = 0; r < 8; r += 2)
              *(u32*)(vT + d * 72 + mb + r) = f2bf2(acc[r], acc[r + 1]);
          }
        }
      }
    }
  }
  LDS_FENCE();

  // ---- stage 4: scores QK^T (block-diagonal 16x16 tiles only) ----
  for (int mt = 0; mt < 4; ++mt) {
    v8f acc;
#pragma unroll
    for (int r = 0; r < 8; ++r) acc[r] = 0.f;
#pragma unroll
    for (int kt = 0; kt < 2; ++kt) {
      v16bf aF = loadAfrag(qRM, 72, mt, kt, lane);
      v16bf bF = loadBfrag(kRM, 72, kt, mt, lane);
      acc = wmma_bf16(aF, bF, acc);
    }
#pragma unroll
    for (int r = 0; r < 8; ++r)
      scF[(mt * 16 + r + 8 * (lane >> 4)) * 17 + (lane & 15)] = acc[r];
  }
  LDS_FENCE();

  // ---- stage 5: softmax (4-token groups) -> prob rows (kRM reused) ----
#pragma unroll
  for (int rr = 0; rr < 2; ++rr) {
    int M = lane * 2 + rr;
    int mt = M >> 4, mr = M & 15;
    int cb = (mr >> 2) << 2;
    float sc[4], mx = -1e30f;
#pragma unroll
    for (int j = 0; j < 4; ++j) {
      sc[j] = scF[M * 17 + cb + j] * 0.125f;
      mx = fmaxf(mx, sc[j]);
    }
    float sum = 0.f;
#pragma unroll
    for (int j = 0; j < 4; ++j) { sc[j] = __expf(sc[j] - mx); sum += sc[j]; }
    float inv = 1.f / sum;
    u32* prow = (u32*)(kRM + M * 72);
#pragma unroll
    for (int q2 = 0; q2 < 32; ++q2) prow[q2] = 0;
    int pb = (mt * 16 + cb) >> 1;
    prow[pb]     = f2bf2(sc[0] * inv, sc[1] * inv);
    prow[pb + 1] = f2bf2(sc[2] * inv, sc[3] * inv);
  }
  LDS_FENCE();

  // ---- stage 6: probs x V -> msa rows (qRM reused); probs resident ----
  {
    v16bf aP[8];
#pragma unroll
    for (int mt = 0; mt < 4; ++mt)
#pragma unroll
      for (int kt = 0; kt < 2; ++kt)
        aP[mt * 2 + kt] = loadAfrag(kRM, 72, mt, kt, lane);
    for (int ct = 0; ct < 4; ++ct) {
      v16bf bV0 = loadBfrag(vT, 72, 0, ct, lane);
      v16bf bV1 = loadBfrag(vT, 72, 1, ct, lane);
      int d = ct * 16 + (lane & 15);
#pragma unroll
      for (int mt = 0; mt < 4; ++mt) {
        v8f acc;
#pragma unroll
        for (int r = 0; r < 8; ++r) acc[r] = 0.f;
        acc = wmma_bf16(aP[mt * 2], bV0, acc);
        acc = wmma_bf16(aP[mt * 2 + 1], bV1, acc);
        int mb = mt * 16 + 8 * (lane >> 4);
#pragma unroll
        for (int r = 0; r < 8; ++r) qRM[(mb + r) * 72 + d] = f2bf(acc[r]);
      }
    }
  }
  LDS_FENCE();

  // ---- stage 7: out-proj + residual -> x rows bf16 (kRM reused) ----
  {
    const u16* Bo = Bqkvo + 3 * 4096;
    v16bf aM[8];
#pragma unroll
    for (int mt = 0; mt < 4; ++mt)
#pragma unroll
      for (int kt = 0; kt < 2; ++kt)
        aM[mt * 2 + kt] = loadAfrag(qRM, 72, mt, kt, lane);
    for (int ct = 0; ct < 4; ++ct) {
      v16bf bF0 = loadBfragG(Bo, ct, lane);
      v16bf bF1 = loadBfragG(Bo, 4 + ct, lane);
      float bias = bo[ct * 16 + (lane & 15)];
      int d = ct * 16 + (lane & 15);
#pragma unroll
      for (int mt = 0; mt < 4; ++mt) {
        v8f acc;
#pragma unroll
        for (int r = 0; r < 8; ++r) acc[r] = bias;
        acc = wmma_bf16(aM[mt * 2], bF0, acc);
        acc = wmma_bf16(aM[mt * 2 + 1], bF1, acc);
        int mb = mt * 16 + 8 * (lane >> 4);
#pragma unroll
        for (int r = 0; r < 8; ++r) {
          int M = mb + r;
          float xv = acc[r] + bf2f(tRM[M * 72 + d]);  // residual with t
          kRM[M * 72 + d] = f2bf(xv);
        }
      }
    }
  }
  LDS_FENCE();

  // ---- stage 8: LayerNorm + heads (row in 32 packed regs) ----
#pragma unroll
  for (int rr = 0; rr < 2; ++rr) {
    int M = lane * 2 + rr;
    const u16* xr = kRM + M * 72;
    u32 rowv[32];
#pragma unroll
    for (int i = 0; i < 32; ++i) rowv[i] = *(const u32*)(xr + 2 * i);
    float mu = 0.f;
#pragma unroll
    for (int i = 0; i < 32; ++i) mu += bflo(rowv[i]) + bfhi(rowv[i]);
    mu *= (1.f / 64.f);
    float var = 0.f;
#pragma unroll
    for (int i = 0; i < 32; ++i) {
      float a = bflo(rowv[i]) - mu;
      float c = bfhi(rowv[i]) - mu;
      var += a * a + c * c;
    }
    var *= (1.f / 64.f);
    float rstd = rsqrtf(var + 1e-5f);
#pragma unroll
    for (int i = 0; i < 32; ++i) {
      float a = (bflo(rowv[i]) - mu) * rstd * ln_g[2 * i] + ln_b[2 * i];
      float c = (bfhi(rowv[i]) - mu) * rstd * ln_g[2 * i + 1] + ln_b[2 * i + 1];
      rowv[i] = f2bf2(a, c);
    }
    int s = M & 3;
    int n = n0 + (M >> 2);
    int r0 = (s == 0) ? 0 : (s == 1) ? 13 : (s == 2) ? 18 : 19;
    int r1 = (s == 0) ? 13 : (s == 1) ? 18 : (s == 2) ? 19 : 25;
    for (int orow = r0; orow < r1; ++orow) {
      float a = bhead[orow];
      const float* wr = Whead + orow * 64;
#pragma unroll
      for (int i = 0; i < 32; ++i)
        a += bflo(rowv[i]) * wr[2 * i] + bfhi(rowv[i]) * wr[2 * i + 1];
      out[((size_t)b * 25 + orow) * 4096 + n] = a;
    }
  }
}

// ---------- host launch ----------
extern "C" void kernel_launch(void* const* d_in, const int* in_sizes, int n_in,
                              void* d_out, int out_size, void* d_ws, size_t ws_size,
                              hipStream_t stream) {
  (void)in_sizes; (void)n_in; (void)out_size; (void)ws_size;
  const float* vp   = (const float*)d_in[0];
  const float* w_af = (const float*)d_in[1];  const float* b_af = (const float*)d_in[2];
  const float* w_df = (const float*)d_in[3];  const float* b_df = (const float*)d_in[4];
  const float* w_wf = (const float*)d_in[5];  const float* b_wf = (const float*)d_in[6];
  const float* w_sf = (const float*)d_in[7];  const float* b_sf = (const float*)d_in[8];
  const float* wq = (const float*)d_in[9];    const float* bq = (const float*)d_in[10];
  const float* wk = (const float*)d_in[11];   const float* bk = (const float*)d_in[12];
  const float* wv = (const float*)d_in[13];   const float* bv = (const float*)d_in[14];
  const float* wo = (const float*)d_in[15];   const float* bo = (const float*)d_in[16];
  const float* ln_g = (const float*)d_in[17]; const float* ln_b = (const float*)d_in[18];
  const float* w_angle = (const float*)d_in[19]; const float* b_angle = (const float*)d_in[20];
  const float* w_depth = (const float*)d_in[21]; const float* b_depth = (const float*)d_in[22];
  const float* w_width = (const float*)d_in[23]; const float* b_width = (const float*)d_in[24];
  const float* w_score = (const float*)d_in[25]; const float* b_score = (const float*)d_in[26];

  char* ws = (char*)d_ws;
  u16*   Bproj    = (u16*)ws;                 // 131072 B
  u16*   Bqkvo    = (u16*)(ws + 131072);      //  32768 B
  float* bias_all = (float*)(ws + 163840);    //   1024 B
  float* Whead    = (float*)(ws + 164864);    //   6400 B
  float* bhead    = (float*)(ws + 171264);    //    128 B

  pack_proj_kernel<<<256, 256, 0, stream>>>(w_af, w_df, w_wf, w_sf, Bproj);
  pack_qkvo_kernel<<<64, 256, 0, stream>>>(wq, wk, wv, wo, Bqkvo);
  pack_small_kernel<<<8, 256, 0, stream>>>(b_af, b_df, b_wf, b_sf,
                                           w_angle, w_depth, w_width, w_score,
                                           b_angle, b_depth, b_width, b_score,
                                           bias_all, Whead, bhead);

  hipFuncSetAttribute(reinterpret_cast<const void*>(li_main_kernel),
                      hipFuncAttributeMaxDynamicSharedMemorySize, SMEM_BYTES);
  const int tiles = 4096;
  const int blocks = (tiles + WAVES_PER_WG - 1) / WAVES_PER_WG;  // 586
  li_main_kernel<<<blocks, 224, SMEM_BYTES, stream>>>(
      vp, Bproj, Bqkvo, bias_all, bq, bk, bv, bo, ln_g, ln_b, Whead, bhead,
      (float*)d_out);
}
